// InfoNCELoss_43499428774286
// MI455X (gfx1250) — matrix operand order repaired and checked by reading, not compiled, not run on hardware
//
#include <hip/hip_runtime.h>

typedef __attribute__((ext_vector_type(16))) _Float16 v16h;
typedef __attribute__((ext_vector_type(8)))  _Float16 v8h;
typedef __attribute__((ext_vector_type(8)))  float    v8f;
typedef __attribute__((ext_vector_type(4)))  float    v4f;
typedef int v4i __attribute__((vector_size(16)));   // matches async-builtin param type

#define TWO_N   8192
#define NHALF   4096
#define DIM     512
#define INV_T   2.0f      // 1 / TEMPERATURE

#define M_BLOCK 64        // rows per workgroup (4 waves x 16)
#define N_BLK   32        // columns staged in LDS per stage
#define NSPLIT  2         // split N range across WGs; row sums are additive
#define N_RANGE (TWO_N / NSPLIT)
#define LDS_STRIDE (DIM + 8)   // +8 halves (16B) pad -> conflict-free b128 reads

// CDNA5 async global->LDS staging if the toolchain exposes the builtins;
// otherwise fall back to the proven load/ds_store staging path.
#if defined(__AMDGCN__) && \
    __has_builtin(__builtin_amdgcn_global_load_async_to_lds_b128) && \
    __has_builtin(__builtin_amdgcn_s_wait_asynccnt)
#define USE_ASYNC 1
#else
#define USE_ASYNC 0
#endif

// ---------------- kernel 0: zero accumulators (graph-safe, deterministic) ----
__global__ void k_init(float* __restrict__ rowSum, float* __restrict__ out) {
  int i = blockIdx.x * blockDim.x + threadIdx.x;
  if (i < TWO_N) rowSum[i] = 0.0f;
  if (i == 0) out[0] = 0.0f;
}

// ---------------- kernel 1: L2-normalize rows, emit f16 reps [8192 x 512] ----
__global__ __launch_bounds__(256) void k_norm(const float* __restrict__ zi,
                                              const float* __restrict__ zj,
                                              _Float16* __restrict__ reps) {
  const int wave = threadIdx.x >> 5;
  const int lane = threadIdx.x & 31;
  const int row  = blockIdx.x * 8 + wave;                  // one wave per row
  const float* src = (row < NHALF) ? (zi + (size_t)row * DIM)
                                   : (zj + (size_t)(row - NHALF) * DIM);
  const v4f* s4 = (const v4f*)(src + lane * 16);
  float vals[16];
#pragma unroll
  for (int i = 0; i < 4; ++i) {
    v4f x = s4[i];
    vals[i*4+0] = x[0]; vals[i*4+1] = x[1]; vals[i*4+2] = x[2]; vals[i*4+3] = x[3];
  }
  float ss = 0.0f;
#pragma unroll
  for (int k = 0; k < 16; ++k) ss += vals[k] * vals[k];
#pragma unroll
  for (int off = 1; off < 32; off <<= 1) ss += __shfl_xor(ss, off, 32);
  const float scale = 1.0f / fmaxf(sqrtf(ss), 1e-12f);
  v8h o0, o1;
#pragma unroll
  for (int k = 0; k < 8; ++k) {
    o0[k] = (_Float16)(vals[k]     * scale);
    o1[k] = (_Float16)(vals[8 + k] * scale);
  }
  _Float16* dst = reps + (size_t)row * DIM + lane * 16;
  *(v8h*)dst       = o0;
  *(v8h*)(dst + 8) = o1;
}

// ---------------- kernel 2: fused sim-GEMM + streaming exp-sum --------------
// Each wave: 16-row strip, A fragments resident in VGPRs for all K.
// B block staged in padded LDS (double-buffered via async copies when
// available). No in-loop masking and no online max: |sim| <= 1 for unit
// rows, and the diagonal term exp(2*(selfdot-1)) is subtracted in k_final.
__global__ __launch_bounds__(128, 1) void k_simloss(const _Float16* __restrict__ reps,
                                                    float* __restrict__ rowSum) {
#if USE_ASYNC
  __shared__ __align__(16) _Float16 ldsB[2][N_BLK][LDS_STRIDE];
#else
  __shared__ __align__(16) _Float16 ldsB[1][N_BLK][LDS_STRIDE];
#endif
  const int wave = threadIdx.x >> 5;
  const int lane = threadIdx.x & 31;
  const int l15  = lane & 15;
  const int sel  = lane >> 4;                       // 0: lanes 0-15, 1: 16-31
  const int mBase   = (int)(blockIdx.x >> 1) * M_BLOCK;
  const int nStart  = (int)(blockIdx.x & 1) * N_RANGE;
  const int rowBase = mBase + wave * 16;

  // A fragments: 16 K-chunks of the wave's 16 rows, ISA 16-bit A layout:
  // halves[0..7] = K k0 + sel*8, halves[8..15] = K k0+16 + sel*8.
  v16h aFrag[16];
  {
    const _Float16* repA = reps + (size_t)(rowBase + l15) * DIM + sel * 8;
#pragma unroll
    for (int kc = 0; kc < 16; ++kc) {
      v8h lo = *(const v8h*)(repA + kc * 32);
      v8h hi = *(const v8h*)(repA + kc * 32 + 16);
      aFrag[kc] = __builtin_shufflevector(lo, hi,
                    0,1,2,3,4,5,6,7,8,9,10,11,12,13,14,15);
    }
  }

  float s[8];
#pragma unroll
  for (int r = 0; r < 8; ++r) s[r] = 0.0f;

#if USE_ASYNC
  // Async stage: 32 rows x 512 halves (32 KB) as 2048 16B chunks, no VGPR transit.
  auto stageAsync = [&](int buf, int n0) {
#pragma unroll
    for (int i = 0; i < 16; ++i) {
      int chunk = (int)threadIdx.x + i * 128;       // 0..2047
      int col   = chunk >> 6;                       // 64 chunks per row
      int kk    = (chunk & 63) * 8;
      __builtin_amdgcn_global_load_async_to_lds_b128(
          (__attribute__((address_space(1))) v4i*)
              (reps + (size_t)(n0 + col) * DIM + kk),
          (__attribute__((address_space(3))) v4i*)&ldsB[buf][col][kk],
          0, 0);
    }
  };
  stageAsync(0, nStart);
  __builtin_amdgcn_s_wait_asynccnt(0);
  __syncthreads();
#endif

  int ib = 0;
  for (int nb = 0; nb < N_RANGE; nb += N_BLK) {
#if USE_ASYNC
    if (nb + N_BLK < N_RANGE)                       // overlap next stage w/ compute
      stageAsync(ib ^ 1, nStart + nb + N_BLK);
#else
    __syncthreads();
#pragma unroll 4
    for (int i = 0; i < 16; ++i) {
      int chunk = (int)threadIdx.x + i * 128;
      int col   = chunk >> 6;
      int kk    = (chunk & 63) * 8;
      v8h vv = *(const v8h*)(reps + (size_t)(nStart + nb + col) * DIM + kk);
      *(v8h*)&ldsB[0][col][kk] = vv;
    }
    __syncthreads();
    if (nb + N_BLK < N_RANGE) {                     // hint next block into cache
      __builtin_prefetch(reps + (size_t)(nStart + nb + N_BLK + (threadIdx.x >> 2)) * DIM
                              + (threadIdx.x & 3) * 128, 0, 3);
    }
#endif
    // Both 16x16 N-tiles interleaved; next-K B fragments preloaded so the
    // ds_loads run a full iteration ahead of their consuming WMMA.
    {
      v8f c0 = {0.f,0.f,0.f,0.f,0.f,0.f,0.f,0.f};
      v8f c1 = {0.f,0.f,0.f,0.f,0.f,0.f,0.f,0.f};
      const _Float16* bP0 = &ldsB[ib][l15][sel * 8];
      const _Float16* bP1 = &ldsB[ib][16 + l15][sel * 8];
      v8h l0 = *(const v8h*)(bP0);  v8h h0 = *(const v8h*)(bP0 + 16);
      v8h l1 = *(const v8h*)(bP1);  v8h h1 = *(const v8h*)(bP1 + 16);
#pragma unroll
      for (int kc = 0; kc < 16; ++kc) {
        v8h nl0, nh0, nl1, nh1;
        if (kc < 15) {
          nl0 = *(const v8h*)(bP0 + (kc + 1) * 32);
          nh0 = *(const v8h*)(bP0 + (kc + 1) * 32 + 16);
          nl1 = *(const v8h*)(bP1 + (kc + 1) * 32);
          nh1 = *(const v8h*)(bP1 + (kc + 1) * 32 + 16);
        }
        v16h b0 = __builtin_shufflevector(l0, h0,
                    0,1,2,3,4,5,6,7,8,9,10,11,12,13,14,15);
        v16h b1 = __builtin_shufflevector(l1, h1,
                    0,1,2,3,4,5,6,7,8,9,10,11,12,13,14,15);
        c0 = __builtin_amdgcn_wmma_f32_16x16x32_f16(false, aFrag[kc], false, b0,
                                                    (short)0, c0, false, false);
        c1 = __builtin_amdgcn_wmma_f32_16x16x32_f16(false, aFrag[kc], false, b1,
                                                    (short)0, c1, false, false);
        if (kc < 15) { l0 = nl0; h0 = nh0; l1 = nl1; h1 = nh1; }
      }
#pragma unroll
      for (int r = 0; r < 8; ++r) {
        s[r] += __expf((c0[r] - 1.0f) * INV_T)      // logits shifted by max = 1/T
              + __expf((c1[r] - 1.0f) * INV_T);     // diagonal removed in k_final
      }
    }
#if USE_ASYNC
    __builtin_amdgcn_s_wait_asynccnt(0);            // next buffer resident in LDS
    __syncthreads();                                // publish to all waves
    ib ^= 1;
#endif
  }
  // Reduce each row over its 16 lanes (xor masks 1..8 stay inside the half).
#pragma unroll
  for (int r = 0; r < 8; ++r) {
#pragma unroll
    for (int off = 1; off < 16; off <<= 1) s[r] += __shfl_xor(s[r], off, 32);
  }
  if (l15 == 0) {                                   // lanes 0 and 16
    const int r0 = rowBase + sel * 8;
#pragma unroll
    for (int r = 0; r < 8; ++r) atomicAdd(&rowSum[r0 + r], s[r]);
  }
}

// ---------------- kernel 3: target logit + diagonal fixup -> mean loss ------
__global__ __launch_bounds__(256) void k_final(const _Float16* __restrict__ reps,
                                               const float* __restrict__ rowSum,
                                               float* __restrict__ out) {
  const int wave = threadIdx.x >> 5;
  const int lane = threadIdx.x & 31;
  const int row  = blockIdx.x * 8 + wave;           // one wave per row
  const _Float16* a = reps + (size_t)row * DIM + lane * 16;
  const _Float16* b = reps + (size_t)(row ^ NHALF) * DIM + lane * 16;  // label = r^4096
  v8h a0 = *(const v8h*)a, a1 = *(const v8h*)(a + 8);
  v8h b0 = *(const v8h*)b, b1 = *(const v8h*)(b + 8);
  float t = 0.0f, d = 0.0f;                         // target logit, self-dot
#pragma unroll
  for (int k = 0; k < 8; ++k) {
    t += (float)a0[k] * (float)b0[k] + (float)a1[k] * (float)b1[k];
    d += (float)a0[k] * (float)a0[k] + (float)a1[k] * (float)a1[k];
  }
#pragma unroll
  for (int off = 1; off < 32; off <<= 1) {
    t += __shfl_xor(t, off, 32);
    d += __shfl_xor(d, off, 32);
  }
  if (lane == 0) {
    // Remove masked diagonal term, then loss_r = log(S_r) - 2*(t - 1).
    float S = rowSum[row] - __expf(INV_T * (d - 1.0f));
    float loss = __logf(S) - INV_T * (t - 1.0f);
    atomicAdd(out, loss * (1.0f / (float)TWO_N));
  }
}

extern "C" void kernel_launch(void* const* d_in, const int* in_sizes, int n_in,
                              void* d_out, int out_size, void* d_ws, size_t ws_size,
                              hipStream_t stream) {
  const float* zi = (const float*)d_in[0];
  const float* zj = (const float*)d_in[1];
  float* out = (float*)d_out;

  _Float16* reps   = (_Float16*)d_ws;                                  // 8 MB
  float*    rowSum = (float*)((char*)d_ws + (size_t)TWO_N * DIM * sizeof(_Float16)); // 32 KB

  k_init   <<<(TWO_N + 255) / 256, 256, 0, stream>>>(rowSum, out);
  k_norm   <<<TWO_N / 8,           256, 0, stream>>>(zi, zj, reps);
  k_simloss<<<(TWO_N / M_BLOCK) * NSPLIT, 128, 0, stream>>>(reps, rowSum);
  k_final  <<<TWO_N / 8,           256, 0, stream>>>(reps, rowSum, out);
}
